// GATLayerEdgeSoftmax_82197084111208
// MI455X (gfx1250) — matrix operand order repaired and checked by reading, not compiled, not run on hardware
//
#include <hip/hip_runtime.h>
#include <hip/hip_bf16.h>

// ---------- types ----------
typedef __attribute__((ext_vector_type(16))) _Float16 v16h;
typedef __attribute__((ext_vector_type(8)))  _Float16 v8h;
typedef __attribute__((ext_vector_type(4)))  _Float16 v4h;
typedef __attribute__((ext_vector_type(8)))  float    v8f;

#define DI 64
#define DO 64
#define KDIM 128
#define HSTRIDE 136          // padded row stride (halfs): 272B => bank-conflict-free b128 rows
#define GMAX_BLOCKS 1024
#define EPS 1e-6f

// ---------- k0: zero numerator (d_out) and denominator ----------
__global__ void k_zero(float* __restrict__ out_num, float* __restrict__ den, int N) {
    int total = N * DO;
    for (int i = blockIdx.x * blockDim.x + threadIdx.x; i < total;
         i += gridDim.x * blockDim.x) {
        out_num[i] = 0.f;
        if (i < N) den[i] = 0.f;
    }
}

// ---------- k1: p[n] = Ww[0:64].x[n], q[n] = Ww[64:128].x[n] ----------
__global__ void k_pq(const float* __restrict__ x, const float* __restrict__ Ww,
                     float* __restrict__ p, float* __restrict__ q, int N) {
    int n = blockIdx.x * blockDim.x + threadIdx.x;
    if (n >= N) return;
    const float* xr = x + (long)n * DI;
    float s0 = 0.f, s1 = 0.f;
#pragma unroll
    for (int k = 0; k < DI; ++k) {
        float xv = xr[k];
        s0 += Ww[k] * xv;
        s1 += Ww[DI + k] * xv;
    }
    p[n] = s0;
    q[n] = s1;
}

// ---------- block max helper (valid on thread 0) ----------
__device__ inline float block_max(float m) {
#pragma unroll
    for (int off = 16; off; off >>= 1)
        m = fmaxf(m, __shfl_xor(m, off, 32));
    __shared__ float sm[8];
    int lane = threadIdx.x & 31, wv = threadIdx.x >> 5;
    if (lane == 0) sm[wv] = m;
    __syncthreads();
    if (threadIdx.x == 0) {
        int nw = blockDim.x >> 5;
        for (int i = 1; i < nw; ++i) m = fmaxf(m, sm[i]);
    }
    return m;
}

// ---------- k2: per-block max of (p[src]+q[tgt]) ----------
__global__ void k_edge_max(const int* __restrict__ src, const int* __restrict__ tgt,
                           const float* __restrict__ p, const float* __restrict__ q,
                           float* __restrict__ bmax, int E) {
    float m = -3.402823466e38f;
    for (int e = blockIdx.x * blockDim.x + threadIdx.x; e < E;
         e += gridDim.x * blockDim.x)
        m = fmaxf(m, p[src[e]] + q[tgt[e]]);
    m = block_max(m);
    if (threadIdx.x == 0) bmax[blockIdx.x] = m;
}

// ---------- k3: reduce block maxima -> amax ----------
__global__ void k_final_max(const float* __restrict__ bmax, float* __restrict__ amax,
                            int nb) {
    float m = -3.402823466e38f;
    for (int i = threadIdx.x; i < nb; i += blockDim.x) m = fmaxf(m, bmax[i]);
    m = block_max(m);
    if (threadIdx.x == 0) amax[0] = m;
}

// ---------- fragment load: two 16B LDS reads -> v16h ----------
__device__ inline v16h ld_frag(const _Float16* base, int gap) {
    v8h lo = *(const v8h*)base;
    v8h hi = *(const v8h*)(base + gap);
    v16h r;
#pragma unroll
    for (int i = 0; i < 8; ++i) { r[i] = lo[i]; r[8 + i] = hi[i]; }
    return r;
}

// ---------- k4: main fused GAT kernel (WMMA) ----------
// 256 threads = 8 waves; each wave owns 16-edge tiles (grid-stride).
// Tile body is branch-free: tail edges are clamped (idx -> E-1, t -> 0, w -> 0)
// so every scatter atomic is an unconditional global_atomic_add_f32 (a +0.0 for
// tail lanes), avoiding per-element exec-mask save/restore sequences.
__global__ __launch_bounds__(256)
void k_gat_wmma(const float* __restrict__ x, const float* __restrict__ Wf,
                const float* __restrict__ bf,
                const int* __restrict__ src, const int* __restrict__ tgt,
                const float* __restrict__ p, const float* __restrict__ q,
                const float* __restrict__ amax,
                float* __restrict__ out_num, float* __restrict__ den, int E) {
    __shared__ _Float16 sWf[DO * HSTRIDE];        // Wf as f16, [out][k], padded
    __shared__ _Float16 sH[8][16 * HSTRIDE];      // per-wave h tile [edge][k]
    __shared__ float    sWgt[8][16];
    __shared__ int      sTgt[8][16];

    const int tid   = threadIdx.x;
    const int lane  = tid & 31;
    const int wv    = tid >> 5;
    const int lmod  = lane & 15;
    const int lhalf = lane >> 4;          // 0: lanes 0-15, 1: lanes 16-31

    // Stage Wf (f32 -> f16) into LDS once per workgroup.
    for (int i = tid; i < DO * KDIM; i += blockDim.x) {
        int row = i >> 7, col = i & 127;
        sWf[row * HSTRIDE + col] = (_Float16)Wf[i];
    }
    __syncthreads();

    // Persistent B fragments: B[kf][nf], lane holds column nf*16+lmod,
    // K = kf*32 + lhalf*16 .. +15 (16 contiguous halfs, packed 2/VGPR).
    v16h B[4][4];
#pragma unroll
    for (int kf = 0; kf < 4; ++kf)
#pragma unroll
        for (int nf = 0; nf < 4; ++nf)
            B[kf][nf] = ld_frag(&sWf[(nf * 16 + lmod) * HSTRIDE + kf * 32 + lhalf * 16], 8);

    float bfr[4];
#pragma unroll
    for (int nf = 0; nf < 4; ++nf) bfr[nf] = bf[nf * 16 + lmod];
    const float am = amax[0];

    const int ntiles  = (E + 15) >> 4;
    const int wave_id = blockIdx.x * 8 + wv;
    const int nwaves  = gridDim.x * 8;

    for (int tile = wave_id; tile < ntiles; tile += nwaves) {
        const int e0 = tile << 4;

        // ---- cooperative gather: lane -> (edge lmod, endpoint lhalf) ----
        {
            const int e = min(e0 + lmod, E - 1);      // clamp: tail lanes refetch a valid row
            const int idx = (lhalf ? tgt : src)[e];
            const float* xr = x + (long)idx * DI;
            _Float16* dst = &sH[wv][lmod * HSTRIDE + lhalf * DI];
#pragma unroll
            for (int kk = 0; kk < 16; ++kk) {
                float4 v = *(const float4*)(xr + kk * 4);
                v4h h4;
                h4[0] = (_Float16)v.x; h4[1] = (_Float16)v.y;
                h4[2] = (_Float16)v.z; h4[3] = (_Float16)v.w;
                *(v4h*)(dst + kk * 4) = h4;
            }
        }

        // ---- edge softmax weight (lanes 0-15, one edge each), branch-free tail ----
        if (lane < 16) {
            const int  e     = e0 + lane;
            const int  ec    = min(e, E - 1);
            const bool valid = (e < E);
            const int  t     = valid ? tgt[ec] : 0;
            const float w    = valid ? __expf(p[src[ec]] + q[t] - am) : 0.f;
            atomicAdd(&den[t], w);                    // +0.0 for tail lanes: harmless
            sWgt[wv][lane] = w;
            sTgt[wv][lane] = t;
        }

        // Wave-private LDS tile: order cross-lane write->read with split DS counter.
        asm volatile("s_wait_dscnt 0" ::: "memory");

        float wm[8];
        int   om[8];                                  // 32-bit element offsets (N*DO < 2^31)
#pragma unroll
        for (int r = 0; r < 8; ++r) {
            const int m = r + lhalf * 8;              // D-matrix row this lane/VGPR holds
            wm[r] = sWgt[wv][m];
            om[r] = sTgt[wv][m] * DO + lmod;
        }

        // ---- 16 x v_wmma_f32_16x16x32_f16: D[16 edges x 64 outs] ----
        v8f C[4] = {};
#pragma unroll
        for (int kf = 0; kf < 4; ++kf) {
            // A layout (16-bit 16x32): lane row = lmod, halfs 0-15 = K klo..klo+15,
            // klo = kf*32 + lhalf*16 (contiguous, matches B).
            v16h A = ld_frag(&sH[wv][lmod * HSTRIDE + kf * 32 + lhalf * 16], 8);
#pragma unroll
            for (int nf = 0; nf < 4; ++nf)
                C[nf] = __builtin_amdgcn_wmma_f32_16x16x32_f16(
                    false, A, false, B[kf][nf], (short)0, C[nf], false, false);
        }

        // ---- relu + bias, scale by edge weight, unconditional scatter ----
        // 32-bit offsets => saddr-base + vgpr-offset atomics, no 64-bit VALU adds.
#pragma unroll
        for (int r = 0; r < 8; ++r) {
            const int   off = om[r];
            const float w   = wm[r];
#pragma unroll
            for (int nf = 0; nf < 4; ++nf) {
                float v = fmaxf(C[nf][r] + bfr[nf], 0.f) * w;  // == 0 for tail edges
                atomicAdd(out_num + off + nf * 16, v);
            }
        }
    }
}

// ---------- k5: o = num / (den + eps), in place ----------
__global__ void k_div(float* __restrict__ out, const float* __restrict__ den, int N) {
    int total = N * DO;
    for (int i = blockIdx.x * blockDim.x + threadIdx.x; i < total;
         i += gridDim.x * blockDim.x)
        out[i] = out[i] / (den[i >> 6] + EPS);
}

// ---------- launcher ----------
extern "C" void kernel_launch(void* const* d_in, const int* in_sizes, int n_in,
                              void* d_out, int out_size, void* d_ws, size_t ws_size,
                              hipStream_t stream) {
    const float* x   = (const float*)d_in[0];
    const float* Wf  = (const float*)d_in[1];
    const float* bf  = (const float*)d_in[2];
    const float* Ww  = (const float*)d_in[3];
    // d_in[4] = bw: cancels in a - max(a); reference init is zeros anyway.
    const int*   src = (const int*)d_in[5];
    const int*   tgt = (const int*)d_in[6];

    const int N = in_sizes[0] / DI;
    const int E = in_sizes[5];

    float* out  = (float*)d_out;
    float* p    = (float*)d_ws;
    float* q    = p + N;
    float* den  = q + N;
    float* bmax = den + N;
    float* amax = bmax + GMAX_BLOCKS;

    k_zero<<<(N * DO + 255) / 256, 256, 0, stream>>>(out, den, N);
    k_pq<<<(N + 255) / 256, 256, 0, stream>>>(x, Ww, p, q, N);
    k_edge_max<<<GMAX_BLOCKS, 256, 0, stream>>>(src, tgt, p, q, bmax, E);
    k_final_max<<<1, 256, 0, stream>>>(bmax, amax, GMAX_BLOCKS);
    k_gat_wmma<<<1024, 256, 0, stream>>>(x, Wf, bf, src, tgt, p, q, amax, out, den, E);
    k_div<<<(N * DO + 255) / 256, 256, 0, stream>>>(out, den, N);
}
